// SparseAttention_21732534518122
// MI455X (gfx1250) — matrix operand order repaired and checked by reading, not compiled, not run on hardware
//
#include <hip/hip_runtime.h>
#include <hip/hip_bf16.h>
#include <math.h>

// ---------------------------------------------------------------------------
// Types for CDNA5 WMMA (wave32): 16x16x32 bf16 -> f32
// ---------------------------------------------------------------------------
typedef __bf16 bf16_t;
typedef bf16_t v16bf __attribute__((ext_vector_type(16)));
typedef bf16_t v8bf  __attribute__((ext_vector_type(8)));
typedef float  v8f   __attribute__((ext_vector_type(8)));

#define D_MODEL 1536
#define SEQ_LEN 4096
#define NUM_HEADS 12
#define HEAD_DIM 128
#define LN_EPS 1e-5f

// LDS tile row stride in bf16 elements: 32 data + 8 pad = 40 (80B, 20 banks)
// -> 16 rows hit 16 distinct bank groups for ds_load_b128 (no conflicts).
#define LDS_STRIDE 40

// ---------------------------------------------------------------------------
// 1) fp32 -> bf16 elementwise convert
// ---------------------------------------------------------------------------
__global__ __launch_bounds__(256)
void cvt_f32_bf16(const float* __restrict__ in, bf16_t* __restrict__ out, int n) {
    int i = blockIdx.x * 256 + threadIdx.x;
    if (i < n) out[i] = (bf16_t)in[i];
}

// ---------------------------------------------------------------------------
// 2) W[K,N] fp32 -> Wt[N,K] bf16 (tiled transpose through LDS)
// ---------------------------------------------------------------------------
__global__ __launch_bounds__(1024)
void transpose_cvt_bf16(const float* __restrict__ in, bf16_t* __restrict__ out,
                        int K, int N) {
    __shared__ float tile[32][33];
    int n = blockIdx.x * 32 + threadIdx.x;
    int k = blockIdx.y * 32 + threadIdx.y;
    tile[threadIdx.y][threadIdx.x] = in[(size_t)k * N + n];
    __syncthreads();
    int n2 = blockIdx.x * 32 + threadIdx.y;
    int k2 = blockIdx.y * 32 + threadIdx.x;
    out[(size_t)n2 * K + k2] = (bf16_t)tile[threadIdx.x][threadIdx.y];
}

// ---------------------------------------------------------------------------
// 3) WMMA GEMM with LDS double buffering:
//      C[M,N] = A[M,K] @ Wt[N,K]^T + bias[N]    (bf16 in, f32 out)
//    Block = 256 threads = 8 wave32 waves; block tile 128(M) x 128(N).
//    K step 32. Per step: block stages A(128x32) and B(128x32) in LDS,
//    each wave runs 8 x v_wmma_f32_16x16x32_bf16 off ds_load_b128 fragments
//    while the next stage's global loads are in flight in registers.
// ---------------------------------------------------------------------------
__global__ __launch_bounds__(256)
void gemm_bf16_wmma(const bf16_t* __restrict__ A, const bf16_t* __restrict__ Wt,
                    const float* __restrict__ bias, float* __restrict__ C,
                    int M, int N, int K) {
    __shared__ __align__(16) bf16_t As[2][128 * LDS_STRIDE];
    __shared__ __align__(16) bf16_t Bs[2][128 * LDS_STRIDE];

    const int tid  = threadIdx.x;
    const int lane = tid & 31;
    const int wave = tid >> 5;          // 0..7
    const int half = lane >> 4;         // 0/1
    const int l16  = lane & 15;

    const int mBlk = blockIdx.y * 128;
    const int n0   = blockIdx.x * 128;

    // Tile copy plan: 128 rows x 64B = 512 chunks of 16B; thread owns 2 chunks.
    const int r0  = tid >> 2;                  // rows 0..63
    const int cc0 = (tid & 3) * 8;             // elem offset 0/8/16/24
    const int r1  = r0 + 64;                   // rows 64..127
    const int cc1 = cc0;

    const bf16_t* Ag0 = A  + (size_t)(mBlk + r0) * K + cc0;
    const bf16_t* Ag1 = A  + (size_t)(mBlk + r1) * K + cc1;
    const bf16_t* Bg0 = Wt + (size_t)(n0 + r0) * K + cc0;
    const bf16_t* Bg1 = Wt + (size_t)(n0 + r1) * K + cc1;

    const int sA0 = r0 * LDS_STRIDE + cc0;
    const int sA1 = r1 * LDS_STRIDE + cc1;

    v8f acc[8] = {};

    // ---- prologue: stage 0 -> LDS[0]
    {
        v8bf ra0 = *(const v8bf*)(Ag0);
        v8bf ra1 = *(const v8bf*)(Ag1);
        v8bf rb0 = *(const v8bf*)(Bg0);
        v8bf rb1 = *(const v8bf*)(Bg1);
        *(v8bf*)&As[0][sA0] = ra0;
        *(v8bf*)&As[0][sA1] = ra1;
        *(v8bf*)&Bs[0][sA0] = rb0;
        *(v8bf*)&Bs[0][sA1] = rb1;
    }
    __syncthreads();

    const int nk = K >> 5;              // K/32 steps
    for (int kt = 0; kt < nk; ++kt) {
        const int buf = kt & 1;
        v8bf ra0, ra1, rb0, rb1;
        const bool more = (kt + 1 < nk);
        if (more) {
            const int k0 = (kt + 1) << 5;
            if (kt + 2 < nk) {
                __builtin_prefetch(Ag0 + k0 + 32, 0, 3);   // global_prefetch_b8
            }
            ra0 = *(const v8bf*)(Ag0 + k0);
            ra1 = *(const v8bf*)(Ag1 + k0);
            rb0 = *(const v8bf*)(Bg0 + k0);
            rb1 = *(const v8bf*)(Bg1 + k0);
        }

        // ---- fragments from LDS + 8 WMMAs
        const bf16_t* Asb = &As[buf][(wave * 16 + l16) * LDS_STRIDE];
        v8bf alo = *(const v8bf*)(Asb + 8 * half);
        v8bf ahi = *(const v8bf*)(Asb + 16 + 8 * half);
        v16bf afrag = __builtin_shufflevector(alo, ahi,
            0,1,2,3,4,5,6,7,8,9,10,11,12,13,14,15);
#pragma unroll
        for (int t = 0; t < 8; ++t) {
            const bf16_t* Bsb = &Bs[buf][(t * 16 + l16) * LDS_STRIDE];
            v8bf blo = *(const v8bf*)(Bsb + 8 * half);
            v8bf bhi = *(const v8bf*)(Bsb + 16 + 8 * half);
            v16bf bfrag = __builtin_shufflevector(blo, bhi,
                0,1,2,3,4,5,6,7,8,9,10,11,12,13,14,15);
            acc[t] = __builtin_amdgcn_wmma_f32_16x16x32_bf16(
                false, afrag, false, bfrag, (short)0, acc[t], false, false);
        }

        // ---- stash next stage into the other buffer
        if (more) {
            const int nb = buf ^ 1;
            *(v8bf*)&As[nb][sA0] = ra0;
            *(v8bf*)&As[nb][sA1] = ra1;
            *(v8bf*)&Bs[nb][sA0] = rb0;
            *(v8bf*)&Bs[nb][sA1] = rb1;
        }
        __syncthreads();
    }

    // ---- epilogue: C = acc + bias  (row = m0 + r + 8*half, col = n0+t*16+l16)
    const int m0 = mBlk + wave * 16;
#pragma unroll
    for (int t = 0; t < 8; ++t) {
        const int n = n0 + t * 16 + l16;
        const float b = bias[n];
#pragma unroll
        for (int r = 0; r < 8; ++r) {
            const int m = m0 + r + 8 * half;
            C[(size_t)m * N + n] = acc[t][r] + b;
        }
    }
}

// ---------------------------------------------------------------------------
// 4) Banded attention: |q-k| <= 2 -> at most 5 keys/query. One 128-thread
//    block per (q, head). fp32 math, bf16 output (feeds O-proj as A matrix).
// ---------------------------------------------------------------------------
__global__ __launch_bounds__(128)
void band_attn(const float* __restrict__ Q, const float* __restrict__ Km,
               const float* __restrict__ V, bf16_t* __restrict__ outb, int S) {
    const int q = blockIdx.x;
    const int h = blockIdx.y;
    const int d = threadIdx.x;
    const int base = h * HEAD_DIM + d;

    __shared__ float red[128];
    __shared__ float probs[8];

    const float qv = Q[(size_t)q * D_MODEL + base];

    int j0 = q - 2; if (j0 < 0) j0 = 0;
    int j1 = q + 2; if (j1 > S - 1) j1 = S - 1;
    const int cnt = j1 - j0 + 1;

    for (int t = 0; t < cnt; ++t) {
        float kv = Km[(size_t)(j0 + t) * D_MODEL + base];
        __syncthreads();
        red[d] = qv * kv;
        __syncthreads();
        for (int s = 64; s > 0; s >>= 1) {
            if (d < s) red[d] += red[d + s];
            __syncthreads();
        }
        if (d == 0) probs[t] = red[0] * 0.08838834764831845f;  // 1/sqrt(128)
    }
    __syncthreads();

    if (d == 0) {
        float mx = probs[0];
        for (int t = 1; t < cnt; ++t) mx = fmaxf(mx, probs[t]);
        float sum = 0.f;
        for (int t = 0; t < cnt; ++t) { float e = expf(probs[t] - mx); probs[t] = e; sum += e; }
        float inv = 1.f / sum;
        for (int t = 0; t < cnt; ++t) probs[t] *= inv;
    }
    __syncthreads();

    float o = 0.f;
    for (int t = 0; t < cnt; ++t)
        o += probs[t] * V[(size_t)(j0 + t) * D_MODEL + base];

    outb[(size_t)q * D_MODEL + base] = (bf16_t)o;
}

// ---------------------------------------------------------------------------
// 5) Residual + LayerNorm, one 256-thread block per row.
// ---------------------------------------------------------------------------
__global__ __launch_bounds__(256)
void residual_ln(const float* __restrict__ x, const float* __restrict__ y,
                 const float* __restrict__ gamma, const float* __restrict__ beta,
                 float* __restrict__ out) {
    const int row = blockIdx.x;
    const int tid = threadIdx.x;
    __shared__ float ssum[256];
    __shared__ float ssq[256];

    float lsum = 0.f, lsq = 0.f;
    for (int i = tid; i < D_MODEL; i += 256) {
        float h = x[(size_t)row * D_MODEL + i] + y[(size_t)row * D_MODEL + i];
        lsum += h; lsq += h * h;
    }
    ssum[tid] = lsum; ssq[tid] = lsq;
    __syncthreads();
    for (int s = 128; s > 0; s >>= 1) {
        if (tid < s) { ssum[tid] += ssum[tid + s]; ssq[tid] += ssq[tid + s]; }
        __syncthreads();
    }
    const float mean = ssum[0] * (1.f / D_MODEL);
    const float var  = ssq[0] * (1.f / D_MODEL) - mean * mean;
    const float rstd = rsqrtf(var + LN_EPS);

    for (int i = tid; i < D_MODEL; i += 256) {
        float h = x[(size_t)row * D_MODEL + i] + y[(size_t)row * D_MODEL + i];
        out[(size_t)row * D_MODEL + i] = (h - mean) * rstd * gamma[i] + beta[i];
    }
}

// ---------------------------------------------------------------------------
// Launcher
// Inputs: 0:x 1:Wq 2:bq 3:Wk 4:bk 5:Wv 6:bv 7:Wo 8:bo 9:gamma 10:beta
// ---------------------------------------------------------------------------
extern "C" void kernel_launch(void* const* d_in, const int* in_sizes, int n_in,
                              void* d_out, int out_size, void* d_ws, size_t ws_size,
                              hipStream_t stream) {
    const float* x     = (const float*)d_in[0];
    const float* Wq    = (const float*)d_in[1];
    const float* bq    = (const float*)d_in[2];
    const float* Wk    = (const float*)d_in[3];
    const float* bk    = (const float*)d_in[4];
    const float* Wv    = (const float*)d_in[5];
    const float* bv    = (const float*)d_in[6];
    const float* Wo    = (const float*)d_in[7];
    const float* bo    = (const float*)d_in[8];
    const float* gamma = (const float*)d_in[9];
    const float* beta  = (const float*)d_in[10];
    float* out = (float*)d_out;

    const int S = SEQ_LEN, D = D_MODEL;
    const size_t SD = (size_t)S * D;
    const size_t DD = (size_t)D * D;

    char* ws = (char*)d_ws;
    size_t off = 0;
    auto alloc = [&](size_t bytes) { char* p = ws + off; off = (off + bytes + 255) & ~(size_t)255; return p; };
    bf16_t* xb   = (bf16_t*)alloc(SD * sizeof(bf16_t));
    bf16_t* WqT  = (bf16_t*)alloc(DD * sizeof(bf16_t));
    bf16_t* WkT  = (bf16_t*)alloc(DD * sizeof(bf16_t));
    bf16_t* WvT  = (bf16_t*)alloc(DD * sizeof(bf16_t));
    bf16_t* WoT  = (bf16_t*)alloc(DD * sizeof(bf16_t));
    float*  Qf   = (float*)alloc(SD * sizeof(float));
    float*  Kf   = (float*)alloc(SD * sizeof(float));
    float*  Vf   = (float*)alloc(SD * sizeof(float));
    bf16_t* attn = (bf16_t*)alloc(SD * sizeof(bf16_t));
    float*  Yf   = (float*)alloc(SD * sizeof(float));
    (void)ws_size;

    cvt_f32_bf16<<<dim3((unsigned)((SD + 255) / 256)), dim3(256), 0, stream>>>(x, xb, (int)SD);

    dim3 tgrid(D / 32, D / 32), tblk(32, 32);
    transpose_cvt_bf16<<<tgrid, tblk, 0, stream>>>(Wq, WqT, D, D);
    transpose_cvt_bf16<<<tgrid, tblk, 0, stream>>>(Wk, WkT, D, D);
    transpose_cvt_bf16<<<tgrid, tblk, 0, stream>>>(Wv, WvT, D, D);
    transpose_cvt_bf16<<<tgrid, tblk, 0, stream>>>(Wo, WoT, D, D);

    dim3 ggrid(D / 128, S / 128), gblk(256);
    gemm_bf16_wmma<<<ggrid, gblk, 0, stream>>>(xb, WqT, bq, Qf, S, D, D);
    gemm_bf16_wmma<<<ggrid, gblk, 0, stream>>>(xb, WkT, bk, Kf, S, D, D);
    gemm_bf16_wmma<<<ggrid, gblk, 0, stream>>>(xb, WvT, bv, Vf, S, D, D);

    band_attn<<<dim3(S, NUM_HEADS), dim3(128), 0, stream>>>(Qf, Kf, Vf, attn, S);

    gemm_bf16_wmma<<<ggrid, gblk, 0, stream>>>(attn, WoT, bo, Yf, S, D, D);

    residual_ln<<<dim3(S), dim3(256), 0, stream>>>(x, Yf, gamma, beta, out);
}
